// PredRNN_64965675319961
// MI455X (gfx1250) — matrix operand order repaired and checked by reading, not compiled, not run on hardware
//
#include <hip/hip_runtime.h>
#include <math.h>

// ---------------------------------------------------------------------------
// PredRNN forward for MI455X (gfx1250, wave32, WMMA).
// All 5x5 / 1x1 convolutions are implicit GEMMs on v_wmma_f32_16x16x32_bf16.
// Weights are pre-packed to bf16 in A-fragment order once per launch so the
// conv inner loop is: 1 contiguous A vector-load per M-tile + 1 im2col gather
// per chunk + MBLK WMMAs.
// ---------------------------------------------------------------------------

typedef __attribute__((ext_vector_type(16))) __bf16 v16bf;
typedef __attribute__((ext_vector_type(8)))  float  v8f;

#define BB   16      // batch
#define TT   10      // frames
#define TOUT 9       // output timesteps (T-1)
#define NHH  64
#define WSP  32      // patched spatial dim
#define NPIX 1024    // 32*32
#define FCC  16      // patch channels

// ---------------------------------------------------------------------------
// Weight pre-pack: f32 (Cout, Cin, FS, FS) -> bf16 in WMMA A-fragment order.
// dst[((mt*NC + kc)*32 + lane)*16 + j] = w[(mt*16 + lane%16)*Ktot + kc*32 + kmap]
// where kmap = ((j<8)?j:j+8) + (lane/16)*8  (CDNA5 ISA 7.12.2, 16-bit A 16x32).
// ---------------------------------------------------------------------------
__global__ __launch_bounds__(256) void pack_weights(const float* __restrict__ w,
                                                    __bf16* __restrict__ dst,
                                                    int Ktot, int NC, int total)
{
    const int idx = blockIdx.x * 256 + threadIdx.x;
    if (idx >= total) return;
    const int j    = idx & 15;
    const int lane = (idx >> 4) & 31;
    const int tc   = idx >> 9;          // mt*NC + kc
    const int kc   = tc % NC;
    const int mt   = tc / NC;
    const int half = lane >> 4;
    const int l16  = lane & 15;
    const int kk   = kc * 32 + (((j < 8) ? j : j + 8) + half * 8);
    const int co   = mt * 16 + l16;
    const float v  = (kk < Ktot) ? w[(size_t)co * Ktot + kk] : 0.0f;
    dst[idx] = (__bf16)v;
}

// ---------------------------------------------------------------------------
// Implicit-GEMM convolution. One wave32 computes MBLK 16x16 output tiles
// (same 16-pixel N-tile, MBLK consecutive co-tiles), so the im2col B gather
// is amortized over MBLK WMMAs; A comes from the bf16 pack as one 32B load.
//   GEMM: M = Cout, N = B*32*32 = 16384, K = Cin*FS*FS (chunks of 32).
// ---------------------------------------------------------------------------
template<int FS_, int MBLK>
__global__ __launch_bounds__(256) void conv_wmma(const float* __restrict__ in,
                                                 const __bf16* __restrict__ wp,
                                                 float* __restrict__ out,
                                                 int Cin, int Cout)
{
    const int KK   = FS_ * FS_;
    const int Ktot = Cin * KK;
    const int NC   = (Ktot + 31) >> 5;

    const int wave = threadIdx.x >> 5;
    const int lane = threadIdx.x & 31;
    const int tile = blockIdx.x * 8 + wave;
    const int mg   = tile >> 10;          // M group (MBLK tiles)
    const int nt   = tile & 1023;         // N tile (pixel / 16)
    const int half = lane >> 4;
    const int l16  = lane & 15;

    // pixel handled by this lane's B column / D column
    const int pn = nt * 16 + l16;
    const int pb = pn >> 10;
    const int py = (pn >> 5) & 31;
    const int px = pn & 31;

    const float* inb = in + ((size_t)pb * Cin << 10);
    // A-fragment pointer for (mt = mg*MBLK, kc = 0)
    const __bf16* wpl = wp + (((size_t)(mg * MBLK) * NC) << 9) + (lane << 4);

    v8f acc[MBLK];
    #pragma unroll
    for (int mb = 0; mb < MBLK; ++mb) acc[mb] = (v8f){};

    for (int kc = 0; kc < NC; ++kc) {
        // ---- B fragment: im2col gather, 32x16 bf16 layout ----
        v16bf bfrag;
        const int kbase = (kc << 5) + half * 8;
        #pragma unroll
        for (int j = 0; j < 16; ++j) {
            const int kk = kbase + ((j < 8) ? j : j + 8);
            float v = 0.0f;
            if (kk < Ktot) {
                if (FS_ == 1) {
                    v = inb[(kk << 10) + (py << 5) + px];
                } else {
                    const int ci = kk / KK;
                    const int r  = kk - ci * KK;
                    const int ky = r / FS_;
                    const int kx = r - ky * FS_;
                    const int iy = py + ky - (FS_ / 2);
                    const int ix = px + kx - (FS_ / 2);
                    if ((unsigned)iy < (unsigned)WSP && (unsigned)ix < (unsigned)WSP)
                        v = inb[(ci << 10) + (iy << 5) + ix];
                }
            }
            bfrag[j] = (__bf16)v;
        }
        // ---- MBLK A fragments (pre-packed, contiguous) + WMMAs ----
        #pragma unroll
        for (int mb = 0; mb < MBLK; ++mb) {
            const v16bf afrag =
                *(const v16bf*)(wpl + (((size_t)(mb * NC + kc)) << 9));
            acc[mb] = __builtin_amdgcn_wmma_f32_16x16x32_bf16(
                false, afrag, false, bfrag, (short)0, acc[mb], false, false);
        }
    }

    float* outb = out + ((size_t)pb * Cout << 10) + (py << 5) + px;
    #pragma unroll
    for (int mb = 0; mb < MBLK; ++mb) {
        #pragma unroll
        for (int r = 0; r < 8; ++r) {
            const int com = (mg * MBLK + mb) * 16 + r + half * 8;
            outb[(size_t)com << 10] = acc[mb][r];
        }
    }
}

// ---------------------------------------------------------------------------
// Per-sample LayerNorm statistics over (C, 32, 32). grid = B.
// ---------------------------------------------------------------------------
__global__ __launch_bounds__(1024) void ln_stats(const float* __restrict__ x,
                                                 float* __restrict__ stats, int C)
{
    const int b = blockIdx.x;
    const int n = C << 10;
    const float* p = x + ((size_t)b * C << 10);
    float s = 0.0f, s2 = 0.0f;
    for (int i = threadIdx.x; i < n; i += 1024) {
        const float v = p[i];
        s += v; s2 += v * v;
    }
    __shared__ float sh[1024];
    __shared__ float sh2[1024];
    sh[threadIdx.x] = s; sh2[threadIdx.x] = s2;
    __syncthreads();
    for (int off = 512; off > 0; off >>= 1) {
        if ((int)threadIdx.x < off) {
            sh[threadIdx.x]  += sh[threadIdx.x + off];
            sh2[threadIdx.x] += sh2[threadIdx.x + off];
        }
        __syncthreads();
    }
    if (threadIdx.x == 0) {
        const float inv = 1.0f / (float)n;
        const float mu  = sh[0] * inv;
        const float var = sh2[0] * inv - mu * mu;
        stats[2 * b]     = mu;
        stats[2 * b + 1] = rsqrtf(var + 1e-5f);
    }
}

__device__ __forceinline__ float sigm(float x) { return 1.0f / (1.0f + expf(-x)); }

// ---------------------------------------------------------------------------
// Gate math with LayerNorm fused (xc: 7 chunks, hc: 4 chunks, mc: 3 chunks).
// Writes c_new, m_new, mem = [c_new ; m_new], o_pre = o_x + o_h.
// ---------------------------------------------------------------------------
__global__ __launch_bounds__(256) void gates_kernel(
    const float* __restrict__ xc, const float* __restrict__ hc, const float* __restrict__ mc,
    const float* __restrict__ stx, const float* __restrict__ sth, const float* __restrict__ stm,
    const float* __restrict__ gx, const float* __restrict__ bx,
    const float* __restrict__ gh, const float* __restrict__ bh,
    const float* __restrict__ gm, const float* __restrict__ bm,
    float* __restrict__ cbuf, float* __restrict__ mbuf,
    float* __restrict__ mem, float* __restrict__ opre)
{
    const int idx = blockIdx.x * 256 + threadIdx.x;   // B*64*1024 exactly
    const int b   = idx >> 16;
    const int j   = (idx >> 10) & 63;
    const int pos = idx & 1023;

    const float mux = stx[2*b], rsx = stx[2*b+1];
    const float muh = sth[2*b], rsh = sth[2*b+1];
    const float mum = stm[2*b], rsm = stm[2*b+1];

    float xln[7];
    #pragma unroll
    for (int ch = 0; ch < 7; ++ch) {
        const int cch  = ch * 64 + j;
        const int gidx = (cch << 10) + pos;
        const float raw = xc[((b * 448 + cch) << 10) + pos];
        xln[ch] = (raw - mux) * rsx * gx[gidx] + bx[gidx];
    }
    float hln[4];
    #pragma unroll
    for (int ch = 0; ch < 4; ++ch) {
        const int cch  = ch * 64 + j;
        const int gidx = (cch << 10) + pos;
        const float raw = hc[((b * 256 + cch) << 10) + pos];
        hln[ch] = (raw - muh) * rsh * gh[gidx] + bh[gidx];
    }
    float mln[3];
    #pragma unroll
    for (int ch = 0; ch < 3; ++ch) {
        const int cch  = ch * 64 + j;
        const int gidx = (cch << 10) + pos;
        const float raw = mc[((b * 192 + cch) << 10) + pos];
        mln[ch] = (raw - mum) * rsm * gm[gidx] + bm[gidx];
    }

    const float cold = cbuf[idx];
    const float mold = mbuf[idx];

    const float i_t = sigm(xln[0] + hln[0]);
    const float f_t = sigm(xln[1] + hln[1] + 1.0f);
    const float g_t = tanhf(xln[2] + hln[2]);
    const float cn  = f_t * cold + i_t * g_t;

    const float i_tp = sigm(xln[3] + mln[0]);
    const float f_tp = sigm(xln[4] + mln[1] + 1.0f);
    const float g_tp = tanhf(xln[5] + mln[2]);
    const float mn   = f_tp * mold + i_tp * g_tp;

    cbuf[idx] = cn;
    mbuf[idx] = mn;
    mem[((b * 128 + j) << 10) + pos]      = cn;
    mem[((b * 128 + 64 + j) << 10) + pos] = mn;
    opre[idx] = xln[6] + hln[3];
}

// ---------------------------------------------------------------------------
// h_new = sigmoid(o_pre + LN(conv_o)) * tanh(conv11)
// ---------------------------------------------------------------------------
__global__ __launch_bounds__(256) void hnew_kernel(
    const float* __restrict__ oc, const float* __restrict__ sto,
    const float* __restrict__ go, const float* __restrict__ bo,
    const float* __restrict__ opre, const float* __restrict__ c11,
    float* __restrict__ h)
{
    const int idx  = blockIdx.x * 256 + threadIdx.x;  // B*64*1024
    const int b    = idx >> 16;
    const int cpos = idx & 65535;
    const float mu = sto[2*b], rs = sto[2*b+1];
    const float oln = (oc[idx] - mu) * rs * go[cpos] + bo[cpos];
    const float o_t = sigm(opre[idx] + oln);
    h[idx] = o_t * tanhf(c11[idx]);
}

// ---------------------------------------------------------------------------
// inp = mask*frame_patch + (1-mask)*xg   (mask==1 for t < IN_NUM)
// ---------------------------------------------------------------------------
__global__ __launch_bounds__(256) void inp_kernel(
    const float* __restrict__ frames, const float* __restrict__ mask,
    const float* __restrict__ xg, float* __restrict__ inp, int t)
{
    const int idx = blockIdx.x * 256 + threadIdx.x;   // B*16*1024
    const int b  = idx >> 14;
    const int cc = (idx >> 10) & 15;
    const int y  = (idx >> 5) & 31;
    const int x  = idx & 31;
    const int py = cc >> 2, px = cc & 3;
    const float ft = frames[((size_t)(b * TT + t) * 128 + (y * 4 + py)) * 128 + (x * 4 + px)];
    float mt = 1.0f;
    if (t >= 4)
        mt = mask[(((size_t)(b * TT + (t - 4)) * 32 + y) * 32 + x) * 16 + cc];
    inp[idx] = mt * ft + (1.0f - mt) * xg[idx];
}

// ---------------------------------------------------------------------------
// Write output frame t: reshape_patch_back of xg (B,16,32,32) -> (B,9,128,128,1)
// ---------------------------------------------------------------------------
__global__ __launch_bounds__(256) void outw_kernel(const float* __restrict__ xg,
                                                   float* __restrict__ out, int t)
{
    const int idx = blockIdx.x * 256 + threadIdx.x;   // B*128*128
    const int b = idx >> 14;
    const int Y = (idx >> 7) & 127;
    const int X = idx & 127;
    const int y = Y >> 2, py = Y & 3;
    const int x = X >> 2, px = X & 3;
    const int cc = py * 4 + px;
    out[((size_t)(b * TOUT + t) << 14) + (Y << 7) + X] =
        xg[((b * FCC + cc) << 10) + (y << 5) + x];
}

__global__ __launch_bounds__(256) void zero_kernel(float* __restrict__ p, int n)
{
    const int idx = blockIdx.x * 256 + threadIdx.x;
    if (idx < n) p[idx] = 0.0f;
}

// ---------------------------------------------------------------------------
// Host-side orchestration
// ---------------------------------------------------------------------------
extern "C" void kernel_launch(void* const* d_in, const int* in_sizes, int n_in,
                              void* d_out, int out_size, void* d_ws, size_t ws_size,
                              hipStream_t stream)
{
    (void)in_sizes; (void)n_in; (void)out_size; (void)ws_size;
    const float* frames = (const float*)d_in[0];
    const float* mask   = (const float*)d_in[1];
    // params['cells'][i] in insertion order: wx,wh,wm,wo,w11,gx,bx,gh,bh,gm,bm,go,bo
    const float* wlast  = (const float*)d_in[2 + 13 * 4];

    float* ws = (float*)d_ws;
    size_t off = 0;
    auto alloc = [&](size_t n) { float* p = ws + off; off += (n + 15) & ~(size_t)15; return p; };

    float* h[4]; for (int i = 0; i < 4; ++i) h[i] = alloc(1048576);
    float* c[4]; for (int i = 0; i < 4; ++i) c[i] = alloc(1048576);
    float* m   = alloc(1048576);
    float* xg  = alloc(262144);
    const size_t zcount = off;            // h, c, m, xg zeroed each launch
    float* inp  = alloc(1048576);
    float* xc   = alloc(7340032);         // (B,448,32,32)
    float* hc   = alloc(4194304);         // (B,256,32,32)
    float* mc   = alloc(3145728);         // (B,192,32,32)
    float* mem  = alloc(2097152);         // (B,128,32,32)
    float* oc   = alloc(1048576);
    float* c11  = alloc(1048576);
    float* opre = alloc(1048576);
    float* stx  = alloc(32);
    float* sth  = alloc(32);
    float* stm  = alloc(32);
    float* sto  = alloc(32);

    // bf16 weight-pack region (A-fragment order), 32B-aligned
    __bf16* packbase = (__bf16*)(ws + off);
    size_t poff = 0;
    auto palloc = [&](size_t nbf16) { __bf16* p = packbase + poff; poff += nbf16; return p; };
    // pack sizes: (Cout/16) * NC * 512 bf16, NC = ceil(Cin*FS*FS / 32)
    __bf16* wxp[4]; __bf16* whp[4]; __bf16* wmp[4]; __bf16* wop[4]; __bf16* w11p[4];
    for (int i = 0; i < 4; ++i) {
        const int cin = (i == 0) ? FCC : NHH;
        const int ncx = (cin * 25 + 31) / 32;
        wxp[i]  = palloc((size_t)28 * ncx * 512);
        whp[i]  = palloc((size_t)16 * 50 * 512);
        wmp[i]  = palloc((size_t)12 * 50 * 512);
        wop[i]  = palloc((size_t)4 * 100 * 512);
        w11p[i] = palloc((size_t)4 * 4 * 512);
    }
    __bf16* wlastp = palloc((size_t)1 * 2 * 512);

    zero_kernel<<<((int)zcount + 255) / 256, 256, 0, stream>>>(ws, (int)zcount);

    // ---- one-time per-launch weight packs ----
    auto pack = [&](const float* w, __bf16* dst, int Cout, int Cin, int FS) {
        const int Ktot  = Cin * FS * FS;
        const int NC    = (Ktot + 31) / 32;
        const int total = (Cout / 16) * NC * 512;
        pack_weights<<<(total + 255) / 256, 256, 0, stream>>>(w, dst, Ktot, NC, total);
    };
    for (int i = 0; i < 4; ++i) {
        const int cin = (i == 0) ? FCC : NHH;
        pack((const float*)d_in[2 + 13 * i + 0], wxp[i],  448, cin, 5);
        pack((const float*)d_in[2 + 13 * i + 1], whp[i],  256, 64,  5);
        pack((const float*)d_in[2 + 13 * i + 2], wmp[i],  192, 64,  5);
        pack((const float*)d_in[2 + 13 * i + 3], wop[i],  64,  128, 5);
        pack((const float*)d_in[2 + 13 * i + 4], w11p[i], 64,  128, 1);
    }
    pack(wlast, wlastp, FCC, 64, 1);

    for (int t = 0; t < TOUT; ++t) {
        inp_kernel<<<1024, 256, 0, stream>>>(frames, mask, xg, inp, t);

        for (int i = 0; i < 4; ++i) {
            const float* gx  = (const float*)d_in[2 + 13 * i + 5];
            const float* bx  = (const float*)d_in[2 + 13 * i + 6];
            const float* gh  = (const float*)d_in[2 + 13 * i + 7];
            const float* bh  = (const float*)d_in[2 + 13 * i + 8];
            const float* gm  = (const float*)d_in[2 + 13 * i + 9];
            const float* bm  = (const float*)d_in[2 + 13 * i + 10];
            const float* go  = (const float*)d_in[2 + 13 * i + 11];
            const float* bo  = (const float*)d_in[2 + 13 * i + 12];

            const float* xin = (i == 0) ? inp : h[i - 1];
            const int    cin = (i == 0) ? FCC : NHH;

            conv_wmma<5, 4><<<(7 * 1024) / 8, 256, 0, stream>>>(xin, wxp[i], xc, cin, 448);
            conv_wmma<5, 4><<<(4 * 1024) / 8, 256, 0, stream>>>(h[i], whp[i], hc, 64, 256);
            conv_wmma<5, 4><<<(3 * 1024) / 8, 256, 0, stream>>>(m, wmp[i], mc, 64, 192);
            ln_stats<<<BB, 1024, 0, stream>>>(xc, stx, 448);
            ln_stats<<<BB, 1024, 0, stream>>>(hc, sth, 256);
            ln_stats<<<BB, 1024, 0, stream>>>(mc, stm, 192);
            gates_kernel<<<4096, 256, 0, stream>>>(xc, hc, mc, stx, sth, stm,
                                                   gx, bx, gh, bh, gm, bm,
                                                   c[i], m, mem, opre);
            conv_wmma<5, 4><<<(1 * 1024) / 8, 256, 0, stream>>>(mem, wop[i], oc, 128, 64);
            ln_stats<<<BB, 1024, 0, stream>>>(oc, sto, 64);
            conv_wmma<1, 4><<<(1 * 1024) / 8, 256, 0, stream>>>(mem, w11p[i], c11, 128, 64);
            hnew_kernel<<<4096, 256, 0, stream>>>(oc, sto, go, bo, opre, c11, h[i]);
        }

        conv_wmma<1, 1><<<1024 / 8, 256, 0, stream>>>(h[3], wlastp, xg, 64, FCC);
        outw_kernel<<<1024, 256, 0, stream>>>(xg, (float*)d_out, t);
    }
}
